// Encoder_69449621176606
// MI455X (gfx1250) — compile-verified, matrix operand order checked by reading, not run on hardware
//
#include <hip/hip_runtime.h>
#include <stdint.h>

// Problem constants (from reference): B=64, S=512, V=32000, E=512, H=512
namespace {
constexpr int kB = 64;
constexpr int kS = 512;
constexpr int kE = 512;
constexpr int kH = 512;
constexpr int kGJ = 2048;              // 4*H gate columns
constexpr int kBlocks = 16;            // persistent: 16 WGs x 8 waves = 128 tile-waves
constexpr int kThreads = 256;

// Workspace layout (bytes)
constexpr size_t kXgBytes   = (size_t)kS * kGJ * kB * 4;       // 256 MB f32 Xg[t][j][m]
constexpr size_t kXbfOff    = kXgBytes;
constexpr size_t kXbfBytes  = (size_t)kB * kS * kE * 2;        // 32 MB bf16 gathered x
constexpr size_t kHbufOff   = kXbfOff + kXbfBytes;
constexpr size_t kHbufBytes = 2ull * kB * kH * 2;              // 128 KB bf16 h double buffer
constexpr size_t kBitsOff   = kHbufOff + kHbufBytes;
constexpr size_t kBitsBytes = (size_t)kS * 4 * sizeof(unsigned);
constexpr size_t kMaskOff   = kBitsOff + kBitsBytes;
constexpr size_t kMaskBytes = (size_t)kS * 4 * sizeof(unsigned);
constexpr size_t kCntOff    = (kMaskOff + kMaskBytes + 127) & ~(size_t)127;

constexpr int kLdsWBytes    = 128 * 1024;                      // Wh bf16 slices only
constexpr int kLdsMaskBytes = kS * 4 * sizeof(unsigned);       // 8 KB eos masks
constexpr int kLdsBytes     = kLdsWBytes + kLdsMaskBytes;      // 136 KB < 320 KB/WGP
}

typedef __attribute__((ext_vector_type(16))) __bf16 v16bf;
typedef __attribute__((ext_vector_type(8)))  float  v8f;

union BfPack {
  uint32_t u[8];
  uint4    q[2];
  v16bf    v;
};
union FPack {
  float4 q[2];
  v8f    v;
};

__device__ __forceinline__ uint32_t f2bf(float f) {
  uint32_t x = __builtin_bit_cast(uint32_t, f);
  return (x + 0x7FFFu + ((x >> 16) & 1u)) >> 16;   // round-to-nearest-even bf16
}
__device__ __forceinline__ uint32_t pack2(float even_k, float odd_k) {
  return f2bf(even_k) | (f2bf(odd_k) << 16);
}
__device__ __forceinline__ float sigm(float x)   { return 1.0f / (1.0f + __expf(-x)); }
__device__ __forceinline__ float tanh_f(float x) { return 2.0f * sigm(2.0f * x) - 1.0f; }

// ---------------------------------------------------------------------------
// Kernel 1: full-chip parallel. Gather embedding rows -> bf16 in WMMA A-chunk
// order, and harvest the raw EOS bit for each (b, t).
// ---------------------------------------------------------------------------
__global__ __launch_bounds__(256)
void embed_pack(const int* __restrict__ inputs, const float* __restrict__ emb,
                uint4* __restrict__ xbf, unsigned* __restrict__ bits) {
  const int g   = blockIdx.x * 256 + threadIdx.x;      // 16-byte chunk id
  const int row = g >> 6;                              // (m*kS + t)
  const int k8  = (g & 63) * 8;
  const int m   = row >> 9;
  const int t   = row & (kS - 1);
  const int tok = inputs[row];

  const float4* er = (const float4*)(emb + (size_t)tok * kE + k8);
  const float4 q0 = er[0], q1 = er[1];
  BfPack p;
  p.u[0] = pack2(q0.x, q0.y); p.u[1] = pack2(q0.z, q0.w);
  p.u[2] = pack2(q1.x, q1.y); p.u[3] = pack2(q1.z, q1.w);
  xbf[g] = p.q[0];

  if (k8 == 0 && q0.y != 0.0f)                         // x_t[m][EOS_ID=1] != 0
    atomicOr(&bits[t * 4 + (m >> 4)], 1u << (m & 15));
}

// ---------------------------------------------------------------------------
// Kernel 2: tiny sequential prefix-OR. masks[t] = is_eos state BEFORE step t.
// ---------------------------------------------------------------------------
__global__ void eos_prefix(const unsigned* __restrict__ bits,
                           unsigned* __restrict__ masks) {
  const int bt = threadIdx.x;
  if (bt >= 4) return;
  unsigned running = 0u;
  for (int t = 0; t < kS; ++t) {
    masks[t * 4 + bt] = running;
    running |= bits[t * 4 + bt];
  }
}

// ---------------------------------------------------------------------------
// Kernel 3: full-chip WMMA GEMM. Xg[t][j][m] = (x_t @ Wx)[m][j] + b[j] for all
// 512 timesteps at once (the parallelizable half of the LSTM FLOPs).
// Block = 32 gate-columns x 64 timesteps; Wx slice (32KB bf16) staged in LDS.
// 512 blocks x 8 waves, 64 16x16 C-tiles per wave, K=512.
// ---------------------------------------------------------------------------
__global__ __launch_bounds__(256)
void xgate_gemm(const unsigned short* __restrict__ xbf,  // [B][S][E] bf16
                const float* __restrict__ Wx,            // (E,4H)
                const float* __restrict__ bias,          // (4H,)
                float* __restrict__ Xg) {                // [S][4H][B]
  __shared__ uint4 lds[(32 * 1024) / 16];  // [c2:2][kb:16] 1KB B-operand slices

  const int lane = threadIdx.x & 31;
  const int wave = threadIdx.x >> 5;
  const int lo   = lane & 15;
  const int h8   = (lane >> 4) * 8;
  const int cg   = blockIdx.x & 63;        // column group (32 cols)
  const int tg   = blockIdx.x >> 6;        // timestep group (64 steps)

  // Stage bf16 Wx slices in WMMA B-operand lane layout.
  for (int idx = wave; idx < 32; idx += 8) {
    const int c2 = idx >> 4;
    const int kb = idx & 15;
    const int j  = cg * 32 + c2 * 16 + lo;
    const int k0 = kb * 32 + h8;
    BfPack p;
#pragma unroll
    for (int jj = 0; jj < 4; ++jj) {
      const int ka = k0 + 2 * jj;
      const int kc = k0 + 16 + 2 * jj;
      p.u[jj]     = pack2(Wx[(size_t)ka * kGJ + j],       Wx[(size_t)(ka + 1) * kGJ + j]);
      p.u[4 + jj] = pack2(Wx[(size_t)kc * kGJ + j],       Wx[(size_t)(kc + 1) * kGJ + j]);
    }
    uint4* dst = &lds[idx * 64 + lane * 2];
    dst[0] = p.q[0];
    dst[1] = p.q[1];
  }
  __syncthreads();

  const float bv0 = bias[cg * 32 + 0 * 16 + lo];   // per-lane bias, c2 = 0/1
  const float bv1 = bias[cg * 32 + 1 * 16 + lo];

  // 512 tiles per block: [t_local:64][b_tile:4][c2:2], round-robin over waves.
  for (int it = wave; it < 512; it += 8) {
    const int t_local = it >> 3;
    const int b_tile  = (it >> 1) & 3;
    const int c2      = it & 1;
    const int t       = tg * 64 + t_local;
    const int m       = b_tile * 16 + lo;
    const int j       = cg * 32 + c2 * 16 + lo;

    v8f acc;
    const float bv = c2 ? bv1 : bv0;
#pragma unroll
    for (int r = 0; r < 8; ++r) acc[r] = bv;

    const uint4* xr = (const uint4*)xbf + ((size_t)m * kS + t) * 64;
#pragma unroll 4
    for (int kb = 0; kb < 16; ++kb) {
      BfPack a, bm;
      const int i8 = kb * 4 + (h8 >> 3);
      a.q[0] = xr[i8];
      a.q[1] = xr[i8 + 2];
      const uint4* bp = &lds[(c2 * 16 + kb) * 64 + lane * 2];
      bm.q[0] = bp[0]; bm.q[1] = bp[1];
      acc = __builtin_amdgcn_wmma_f32_16x16x32_bf16(
          false, a.v, false, bm.v, (short)0, acc, false, false);
    }

    // Store transposed: rows (m) contiguous so the scan loads float4 pairs.
    float4* op = (float4*)(Xg + ((size_t)t * kGJ + j) * kB + b_tile * 16 + h8);
    op[0] = make_float4(acc[0], acc[1], acc[2], acc[3]);
    op[1] = make_float4(acc[4], acc[5], acc[6], acc[7]);
  }
}

// ---------------------------------------------------------------------------
// Device-scope software grid barrier across the 16 resident workgroups.
// ---------------------------------------------------------------------------
__device__ __forceinline__ void grid_barrier(unsigned* cnt, unsigned target) {
  __syncthreads();
  if (threadIdx.x == 0) {
    __threadfence();  // release h stores
    __hip_atomic_fetch_add(cnt, 1u, __ATOMIC_RELEASE, __HIP_MEMORY_SCOPE_AGENT);
    while (__hip_atomic_load(cnt, __ATOMIC_ACQUIRE, __HIP_MEMORY_SCOPE_AGENT) < target)
      __builtin_amdgcn_s_sleep(1);
  }
  __syncthreads();
}

// ---------------------------------------------------------------------------
// Kernel 4: persistent recurrent scan — only the forced-sequential half
// (h @ Wh, K=512). c/h tiles live in VGPRs for all 512 steps; Wh bf16 slices
// in LDS; Xg accumulator tiles prefetched one step ahead (hidden behind the
// grid barrier); h passes between steps via global bf16 double buffer.
// ---------------------------------------------------------------------------
__global__ __launch_bounds__(kThreads)
void lstm_persistent(const float* __restrict__ Xg,       // [S][4H][B]
                     const float* __restrict__ Wh,       // (H,4H)
                     const unsigned* __restrict__ masks, // [S][4]
                     float* __restrict__ out,            // c | h
                     unsigned short* __restrict__ hbuf,  // [2][B*H] bf16
                     unsigned* __restrict__ cnt) {
  extern __shared__ uint4 ldsq[];   // [n2:2][gate:4][kb:16] 1KB slices, then masks

  const int lane   = threadIdx.x & 31;
  const int wave   = threadIdx.x >> 5;
  const int lo     = lane & 15;
  const int h8     = (lane >> 4) * 8;
  const int b_tile = wave >> 1;
  const int n2     = wave & 1;
  const int b_base = b_tile * 16;
  const int n_glob = blockIdx.x * 32 + n2 * 16 + lo;

  // Stage bf16 Wh slices (K = 0..511 of the recurrent GEMM).
  for (int idx = wave; idx < 128; idx += 8) {
    const int n2f  = idx >> 6;
    const int gate = (idx >> 4) & 3;
    const int kb   = idx & 15;
    const int col  = blockIdx.x * 32 + n2f * 16 + lo;
    const int gcol = gate * 512 + col;
    const int k0   = kb * 32 + h8;
    BfPack p;
#pragma unroll
    for (int jj = 0; jj < 4; ++jj) {
      const int ka = k0 + 2 * jj;
      const int kc = k0 + 16 + 2 * jj;
      p.u[jj]     = pack2(Wh[(size_t)ka * kGJ + gcol], Wh[(size_t)(ka + 1) * kGJ + gcol]);
      p.u[4 + jj] = pack2(Wh[(size_t)kc * kGJ + gcol], Wh[(size_t)(kc + 1) * kGJ + gcol]);
    }
    uint4* dst = &ldsq[idx * 64 + lane * 2];
    dst[0] = p.q[0];
    dst[1] = p.q[1];
  }
  unsigned* ldsmask = (unsigned*)(ldsq + (kLdsWBytes >> 4));
  for (int i = threadIdx.x; i < kS * 4; i += kThreads) ldsmask[i] = masks[i];
  __syncthreads();

  float creg[8], hreg[8];
#pragma unroll
  for (int r = 0; r < 8; ++r) { creg[r] = 0.0f; hreg[r] = 0.0f; }

  const int m = b_base + lo;  // batch row this lane feeds into the A operand

  // Prefetch Xg accumulator tiles for step 0.
  FPack nxt[4];
#pragma unroll
  for (int g = 0; g < 4; ++g) {
    const float4* gp = (const float4*)(Xg + ((size_t)0 * kGJ + (g * 512 + n_glob)) * kB + b_base + h8);
    nxt[g].q[0] = gp[0];
    nxt[g].q[1] = gp[1];
  }

  for (int t = 0; t < kS; ++t) {
    const unsigned short* hold = hbuf + (size_t)(t & 1) * (kB * kH);
    unsigned short*       hnew = hbuf + (size_t)((t + 1) & 1) * (kB * kH);
    const unsigned mask = ldsmask[t * 4 + b_tile];

    v8f acc[4];
#pragma unroll
    for (int g = 0; g < 4; ++g) acc[g] = nxt[g].v;

    // Recurrent half: gates += h @ Wh (K = 512).
    const uint4* hr = (const uint4*)(hold + (size_t)m * kH);
#pragma unroll 4
    for (int kb = 0; kb < 16; ++kb) {
      BfPack a;
      const int i8 = kb * 4 + (h8 >> 3);
      a.q[0] = hr[i8];
      a.q[1] = hr[i8 + 2];
#pragma unroll
      for (int g = 0; g < 4; ++g) {
        BfPack bm;
        const uint4* bp = &ldsq[(((n2 * 4 + g) * 16 + kb) * 64) + lane * 2];
        bm.q[0] = bp[0]; bm.q[1] = bp[1];
        acc[g] = __builtin_amdgcn_wmma_f32_16x16x32_bf16(
            false, a.v, false, bm.v, (short)0, acc[g], false, false);
      }
    }

    // Elementwise LSTM cell + EOS freeze. C-tile: VGPR r = row b_base+h8+r.
#pragma unroll
    for (int r = 0; r < 8; ++r) {
      const float iv = acc[0][r], fv = acc[1][r], gv = acc[2][r], ov = acc[3][r];
      const float nc = sigm(fv) * creg[r] + sigm(iv) * tanh_f(gv);
      const float nh = sigm(ov) * tanh_f(nc);
      const bool keep = (mask >> (h8 + r)) & 1u;
      creg[r] = keep ? creg[r] : nc;
      hreg[r] = keep ? hreg[r] : nh;
      const int row = b_base + h8 + r;
      hnew[(size_t)row * kH + n_glob] = (unsigned short)f2bf(hreg[r]);
    }

    // Prefetch next step's Xg tiles; latency hides behind the grid barrier.
    if (t + 1 < kS) {
#pragma unroll
      for (int g = 0; g < 4; ++g) {
        const float4* gp = (const float4*)(Xg + ((size_t)(t + 1) * kGJ + (g * 512 + n_glob)) * kB + b_base + h8);
        nxt[g].q[0] = gp[0];
        nxt[g].q[1] = gp[1];
      }
    }

    grid_barrier(cnt, (unsigned)(kBlocks * (t + 1)));
  }

  // Final state: out = [c (B,H) | h (B,H)] in f32.
#pragma unroll
  for (int r = 0; r < 8; ++r) {
    const int row = b_base + h8 + r;
    out[(size_t)row * kH + n_glob]                   = creg[r];
    out[(size_t)kB * kH + (size_t)row * kH + n_glob] = hreg[r];
  }
}

extern "C" void kernel_launch(void* const* d_in, const int* in_sizes, int n_in,
                              void* d_out, int out_size, void* d_ws, size_t ws_size,
                              hipStream_t stream) {
  (void)in_sizes; (void)n_in; (void)out_size; (void)ws_size;
  const int*   inputs = (const int*)d_in[0];
  const float* emb    = (const float*)d_in[1];
  const float* Wx     = (const float*)d_in[2];
  const float* Wh     = (const float*)d_in[3];
  const float* bias   = (const float*)d_in[4];
  float*       out    = (float*)d_out;

  char* ws = (char*)d_ws;
  float*          Xg    = (float*)ws;
  uint4*          xbf4  = (uint4*)(ws + kXbfOff);
  unsigned short* xbf16 = (unsigned short*)(ws + kXbfOff);
  unsigned short* hbuf  = (unsigned short*)(ws + kHbufOff);
  unsigned*       bits  = (unsigned*)(ws + kBitsOff);
  unsigned*       masks = (unsigned*)(ws + kMaskOff);
  unsigned*       cnt   = (unsigned*)(ws + kCntOff);

  // Zero the mutable tail (h double buffer, eos bits, counter) every call.
  (void)hipMemsetAsync(ws + kHbufOff, 0, (kCntOff - kHbufOff) + 256, stream);

  // 1) Parallel gather + bf16 pack of x; harvest raw EOS bits.
  const int packThreads = (kB * kS * kE) / 8;
  embed_pack<<<dim3(packThreads / 256), dim3(256), 0, stream>>>(inputs, emb, xbf4, bits);

  // 2) Sequential prefix-OR of EOS bits (trivial).
  eos_prefix<<<dim3(1), dim3(64), 0, stream>>>(bits, masks);

  // 3) Full-chip WMMA GEMM for the parallel half: Xg = x@Wx + b for all t.
  xgate_gemm<<<dim3(512), dim3(256), 0, stream>>>(xbf16, Wx, bias, Xg);

  // 4) Persistent recurrent scan (h@Wh only on the critical path).
  lstm_persistent<<<dim3(kBlocks), dim3(kThreads), kLdsBytes, stream>>>(
      Xg, Wh, masks, out, hbuf, cnt);
}
